// SeG_68169720922361
// MI455X (gfx1250) — compile-verified, matrix operand order checked by reading, not run on hardware
//
#include <hip/hip_runtime.h>
#include <hip/hip_bf16.h>

// ---------------------------------------------------------------------------
// SeG selective-gate + bag-mean + classifier, CDNA5 (gfx1250, wave32)
// WMMA bf16 GEMMs + async-to-LDS double-buffered staging + TDM broadcast
// + hardware tanh activations.
// ---------------------------------------------------------------------------

typedef __bf16 bf16_t;
typedef __attribute__((ext_vector_type(16))) __bf16 v16bf;
typedef __attribute__((ext_vector_type(8)))  __bf16 v8bf;
typedef __attribute__((ext_vector_type(8)))  float  v8f;
typedef __attribute__((ext_vector_type(4)))  unsigned int u32x4;
typedef __attribute__((ext_vector_type(8)))  unsigned int u32x8;

static constexpr int N_SENT = 100000;
static constexpr int DU  = 150;   // 3*WORD_DIM
static constexpr int DUP = 160;   // padded (mult of 32)
static constexpr int DH  = 690;   // 3*HIDDEN
static constexpr int DHP = 704;   // padded (mult of 64)
static constexpr int NB  = 5000;
static constexpr int RN  = 53;
static constexpr int RNP = 64;

// ---- CDNA5 async/TDM primitives -------------------------------------------
__device__ __forceinline__ void async_ld_b128(const bf16_t* g, bf16_t* l) {
    unsigned lds = (unsigned)(unsigned long long)l;       // LDS aperture: low 32 bits
    unsigned long long ga = (unsigned long long)g;
    asm volatile("global_load_async_to_lds_b128 %0, %1, off"
                 :: "v"(lds), "v"(ga) : "memory");
}
__device__ __forceinline__ void wait_async0() {
    asm volatile("s_wait_asynccnt 0x0" ::: "memory");
}
__device__ __forceinline__ void wait_async5() {           // ping-pong depth
    asm volatile("s_wait_asynccnt 0x5" ::: "memory");
}

// TDM: contiguous 1-D block copy global -> LDS (TENSORcnt). One wave issues.
__device__ __forceinline__ void tdm_copy_1d(const bf16_t* g, bf16_t* l,
                                            unsigned nelem) {
    unsigned long long ga  = (unsigned long long)g;
    unsigned           lds = (unsigned)(unsigned long long)l;
    unsigned long long q0g0 = 1ull | ((unsigned long long)lds << 32);
    unsigned long long q1g0 = (ga & 0x01FFFFFFFFFFFFFFull) | (2ull << 62);
    u32x4 g0; g0.x = (unsigned)q0g0; g0.y = (unsigned)(q0g0 >> 32);
              g0.z = (unsigned)q1g0; g0.w = (unsigned)(q1g0 >> 32);
    unsigned long long q0g1 = (1ull << 16)                          // data_size=2B
                            | ((unsigned long long)(nelem & 0xFFFF) << 48);
    unsigned long long q1g1 = (unsigned long long)(nelem >> 16)     // dim0 hi16
                            | (1ull << 16)                          // tensor_dim1=1
                            | ((unsigned long long)(nelem & 0xFFFF) << 48); // tile_dim0
    unsigned long long q2g1 = ((unsigned long long)nelem << 32);    // dim0_stride lo32
    unsigned long long q3g1 = 0ull;
    u32x8 g1; g1.s0 = (unsigned)q0g1; g1.s1 = (unsigned)(q0g1 >> 32);
              g1.s2 = (unsigned)q1g1; g1.s3 = (unsigned)(q1g1 >> 32);
              g1.s4 = (unsigned)q2g1; g1.s5 = (unsigned)(q2g1 >> 32);
              g1.s6 = (unsigned)q3g1; g1.s7 = (unsigned)(q3g1 >> 32);
    asm volatile("tensor_load_to_lds %0, %1" :: "s"(g0), "s"(g1) : "memory");
}

// ---- fast activations (gfx1250 v_tanh_f32 when available) -----------------
__device__ __forceinline__ float tanh_fast(float x) {
#if __has_builtin(__builtin_amdgcn_tanhf)
    return __builtin_amdgcn_tanhf(x);
#elif __has_builtin(__builtin_amdgcn_tanh_f32)
    return __builtin_amdgcn_tanh_f32(x);
#else
    float e = __expf(2.f * x);
    return (e - 1.f) * __builtin_amdgcn_rcpf(e + 1.f);
#endif
}
__device__ __forceinline__ float sigmoid_fast(float x) {
    return 0.5f * tanh_fast(0.5f * x) + 0.5f;
}

// ---- WMMA helpers ---------------------------------------------------------
__device__ __forceinline__ v16bf frag16(const bf16_t* tile, int stride,
                                        int mn, int half, int k0) {
    const bf16_t* p = tile + mn * stride + k0 + half * 8;
    v8bf lo = *reinterpret_cast<const v8bf*>(p);
    v8bf hi = *reinterpret_cast<const v8bf*>(p + 16);
    return __builtin_shufflevector(lo, hi, 0,1,2,3,4,5,6,7,
                                            8,9,10,11,12,13,14,15);
}
__device__ __forceinline__ v8f wmma_bf16(v16bf a, v16bf b, v8f c) {
    return __builtin_amdgcn_wmma_f32_16x16x32_bf16(
        false, a, false, b, (short)0, c, false, false);
}

// --------------------------- prep kernels ----------------------------------
__global__ void k_segids(const int* __restrict__ scope, int* __restrict__ seg) {
    int r = blockIdx.x * 256 + threadIdx.x;
    if (r >= N_SENT) return;
    int lo = 0, hi = NB;                       // first b with ends[b] > r
    while (lo < hi) {
        int mid = (lo + hi) >> 1;
        if (scope[mid * 2 + 1] > r) hi = mid; else lo = mid + 1;
    }
    seg[r] = lo;
}

__global__ void k_zero(float* __restrict__ p, int n) {
    int i = blockIdx.x * 256 + threadIdx.x;
    if (i < n) p[i] = 0.f;
}

__global__ void k_prep_w(const float* __restrict__ W1, const float* __restrict__ W2,
                         const float* __restrict__ Wc,
                         bf16_t* __restrict__ W1t, bf16_t* __restrict__ W2t,
                         bf16_t* __restrict__ Wct) {
    int i = blockIdx.x * 256 + threadIdx.x;
    if (i < DUP * DUP) {                       // W1t[n][k] = W1[k][n]
        int n = i / DUP, k = i % DUP;
        W1t[i] = (bf16_t)((n < DU && k < DU) ? W1[k * DU + n] : 0.f);
    }
    if (i < DHP * DUP) {                       // W2t[n][k] = W2[k][n]
        int n = i / DUP, k = i % DUP;
        W2t[i] = (bf16_t)((n < DH && k < DU) ? W2[k * DH + n] : 0.f);
    }
    if (i < RNP * DHP) {                       // Wct[n][k] = Wc[k][n]
        int n = i / DHP, k = i % DHP;
        Wct[i] = (bf16_t)((n < RN && k < DH) ? Wc[k * RN + n] : 0.f);
    }
}

__global__ void k_u2bf(const float* __restrict__ U, bf16_t* __restrict__ Ubf) {
    long i = (long)blockIdx.x * 256 + threadIdx.x;
    if (i >= (long)N_SENT * DUP) return;
    int r = (int)(i / DUP), k = (int)(i % DUP);
    Ubf[i] = (bf16_t)((k < DU) ? U[(long)r * DU + k] : 0.f);
}

// --------------------------- fc1: H = tanh(U@W1+b1) ------------------------
__global__ void k_fc1(const bf16_t* __restrict__ Ubf, const bf16_t* __restrict__ W1t,
                      const float* __restrict__ b1, bf16_t* __restrict__ H) {
    __shared__ __align__(16) bf16_t sU[32 * DUP];    // 10 KB
    __shared__ __align__(16) bf16_t sW[DUP * DUP];   // 50 KB (n-major W1^T)
    const int tid  = threadIdx.x;
    const int row0 = blockIdx.x * 32;
    const int lane = tid & 31, wave = tid >> 5;
    const int half = lane >> 4, idx = lane & 15;
    const bool full = (row0 + 32 <= N_SENT);

    if (wave == 0) {                                 // TDM broadcast of W1^T
        tdm_copy_1d(W1t, sW, DUP * DUP);
        __builtin_amdgcn_s_wait_tensorcnt(0);
    }
    for (int i = tid; i < (32 * DUP) / 8; i += 256) {
        int elem = i * 8;
        if (row0 + elem / DUP < N_SENT)
            async_ld_b128(Ubf + (long)row0 * DUP + elem, &sU[elem]);
    }
    wait_async0();
    __syncthreads();

    for (int t = wave; t < 20; t += 8) {             // 2 m-tiles x 10 n-tiles
        int mt = t / 10, nt = t % 10;
        v8f acc = {};
        #pragma unroll
        for (int kk = 0; kk < DUP; kk += 32) {
            v16bf a = frag16(sU + mt * 16 * DUP, DUP, idx, half, kk);
            v16bf b = frag16(sW + nt * 16 * DUP, DUP, idx, half, kk);
            acc = wmma_bf16(a, b, acc);
        }
        int n = nt * 16 + idx;
        float bias = (n < DU) ? b1[n] : 0.f;
        if (full) {
            #pragma unroll
            for (int i = 0; i < 8; ++i) {
                int gr = row0 + mt * 16 + half * 8 + i;
                H[(long)gr * DUP + n] = (bf16_t)tanh_fast(acc[i] + bias);
            }
        } else {
            #pragma unroll
            for (int i = 0; i < 8; ++i) {
                int gr = row0 + mt * 16 + half * 8 + i;
                if (gr < N_SENT)
                    H[(long)gr * DUP + n] = (bf16_t)tanh_fast(acc[i] + bias);
            }
        }
    }
}

// ------------- fc2 + sigmoid gate + segment-sum atomics --------------------
// Block: 64 rows; W2^T chunks double-buffered via async-to-LDS (in-order
// ASYNCcnt retirement: wait<=5 guarantees the previous chunk has landed).
__global__ void k_fc2_gate(const bf16_t* __restrict__ H,
                           const bf16_t* __restrict__ W2t,
                           const float* __restrict__ b2,
                           const float* __restrict__ S,
                           const int*   __restrict__ seg,
                           float* __restrict__ bagsum) {
    __shared__ __align__(16) bf16_t sH[64 * DUP];       // 20 KB
    __shared__ __align__(16) bf16_t sW[2][64 * DUP];    // 2 x 20 KB ping-pong
    const int tid  = threadIdx.x;
    const int row0 = blockIdx.x * 64;
    const int lane = tid & 31, wave = tid >> 5;
    const int half = lane >> 4, idx = lane & 15;
    const bool full = (row0 + 64 <= N_SENT);
    constexpr int NCH = DHP / 64;                       // 11 chunks

    for (int i = tid; i < (64 * DUP) / 8; i += 256) {   // async H tile (5/wave)
        int elem = i * 8;
        if (row0 + elem / DUP < N_SENT)
            async_ld_b128(H + (long)row0 * DUP + elem, &sH[elem]);
    }
    for (int i = tid; i < (64 * DUP) / 8; i += 256)     // chunk 0 (5/wave)
        async_ld_b128(W2t + i * 8, &sW[0][i * 8]);

    for (int c = 0; c < NCH; ++c) {
        if (c + 1 < NCH) {                              // prefetch next chunk
            for (int i = tid; i < (64 * DUP) / 8; i += 256)
                async_ld_b128(W2t + (long)(c + 1) * 64 * DUP + i * 8,
                              &sW[(c + 1) & 1][i * 8]);
            wait_async5();                              // chunk c (and sH) done
        } else {
            wait_async0();
        }
        __syncthreads();

        const bf16_t* sWc = sW[c & 1];
        const int ncol0 = c * 64;
        for (int t = wave; t < 16; t += 8) {            // 4 m-tiles x 4 n-tiles
            int mt = t >> 2, nt = t & 3;
            v8f acc = {};
            #pragma unroll
            for (int kk = 0; kk < DUP; kk += 32) {
                v16bf a = frag16(sH  + mt * 16 * DUP, DUP, idx, half, kk);
                v16bf b = frag16(sWc + nt * 16 * DUP, DUP, idx, half, kk);
                acc = wmma_bf16(a, b, acc);
            }
            int n = ncol0 + nt * 16 + idx;
            if (n < DH) {
                float bias = b2[n];
                if (full) {
                    #pragma unroll
                    for (int i = 0; i < 8; ++i) {
                        int gr = row0 + mt * 16 + half * 8 + i;
                        float g = sigmoid_fast(acc[i] + bias);
                        float x = g * S[(long)gr * DH + n];
                        atomicAdd(&bagsum[(long)seg[gr] * DH + n], x);
                    }
                } else {
                    #pragma unroll
                    for (int i = 0; i < 8; ++i) {
                        int gr = row0 + mt * 16 + half * 8 + i;
                        if (gr < N_SENT) {
                            float g = sigmoid_fast(acc[i] + bias);
                            float x = g * S[(long)gr * DH + n];
                            atomicAdd(&bagsum[(long)seg[gr] * DH + n], x);
                        }
                    }
                }
            }
        }
        __syncthreads();                                // reads done before overwrite
    }
}

// --------------------------- bag mean -> bf16 ------------------------------
__global__ void k_bagmean(const float* __restrict__ bagsum,
                          const int* __restrict__ scope,
                          bf16_t* __restrict__ Bbf) {
    int i = blockIdx.x * 256 + threadIdx.x;
    if (i >= NB * DHP) return;
    int b = i / DHP, n = i % DHP;
    float v = 0.f;
    if (n < DH) {
        float cnt = fmaxf((float)(scope[b * 2 + 1] - scope[b * 2]), 1.f);
        v = bagsum[b * DH + n] / cnt;
    }
    Bbf[i] = (bf16_t)v;
}

// --------------------------- classifier: B @ Wc + bc -----------------------
static constexpr int KC = 352;                      // DHP/2, mult of 32
__global__ void k_cls(const bf16_t* __restrict__ Bbf,
                      const bf16_t* __restrict__ Wct,
                      const float* __restrict__ bc,
                      float* __restrict__ out) {
    __shared__ __align__(16) bf16_t sW[RNP * KC];   // 44 KB
    const int tid  = threadIdx.x;
    const int row0 = blockIdx.x * 32;
    const int lane = tid & 31, wave = tid >> 5;
    const int half = lane >> 4, idx = lane & 15;
    const int mt = wave >> 2, nt = wave & 3;

    v8f acc = {};
    for (int c = 0; c < DHP / KC; ++c) {
        __syncthreads();
        for (int i = tid; i < (RNP * KC) / 8; i += 256) {
            int elem = i * 8;
            int nl = elem / KC, kl = elem % KC;
            async_ld_b128(Wct + (long)nl * DHP + c * KC + kl, &sW[elem]);
        }
        wait_async0();
        __syncthreads();

        int gr = row0 + mt * 16 + idx;              // A row for this lane
        #pragma unroll
        for (int kk = 0; kk < KC; kk += 32) {
            v16bf a = {};
            if (gr < NB) {
                const bf16_t* p = Bbf + (long)gr * DHP + c * KC + kk + half * 8;
                v8bf lo = *reinterpret_cast<const v8bf*>(p);
                v8bf hi = *reinterpret_cast<const v8bf*>(p + 16);
                a = __builtin_shufflevector(lo, hi, 0,1,2,3,4,5,6,7,
                                                     8,9,10,11,12,13,14,15);
            }
            v16bf b = frag16(sW + nt * 16 * KC, KC, idx, half, kk);
            acc = wmma_bf16(a, b, acc);
        }
    }

    int n = nt * 16 + idx;
    if (n < RN) {
        float bias = bc[n];
        #pragma unroll
        for (int i = 0; i < 8; ++i) {
            int gr = row0 + mt * 16 + half * 8 + i;
            if (gr < NB) out[gr * RN + n] = acc[i] + bias;
        }
    }
}

// ---------------------------------------------------------------------------
extern "C" void kernel_launch(void* const* d_in, const int* in_sizes, int n_in,
                              void* d_out, int out_size, void* d_ws, size_t ws_size,
                              hipStream_t stream) {
    (void)in_sizes; (void)n_in; (void)out_size; (void)ws_size;
    const float* S     = (const float*)d_in[0];
    const float* U     = (const float*)d_in[1];
    const int*   scope = (const int*)  d_in[2];
    const float* W1    = (const float*)d_in[3];
    const float* b1    = (const float*)d_in[4];
    const float* W2    = (const float*)d_in[5];
    const float* b2    = (const float*)d_in[6];
    const float* Wc    = (const float*)d_in[7];
    const float* bc    = (const float*)d_in[8];
    float* out = (float*)d_out;

    char* ws = (char*)d_ws;
    bf16_t* Ubf    = (bf16_t*)(ws);                 // N*160 bf16   = 32.0 MB
    bf16_t* H      = (bf16_t*)(ws + 32000000);      // N*160 bf16   = 32.0 MB
    int*    seg    = (int*)   (ws + 64000000);      // N int        =  0.4 MB
    float*  bagsum = (float*) (ws + 64400128);      // NB*690 f32   = 13.8 MB
    bf16_t* Bbf    = (bf16_t*)(ws + 78200320);      // NB*704 bf16  =  7.0 MB
    bf16_t* W1t    = (bf16_t*)(ws + 85240320);      // 160*160 bf16 = 50 KB
    bf16_t* W2t    = (bf16_t*)(ws + 85291520);      // 704*160 bf16 = 220 KB
    bf16_t* Wct    = (bf16_t*)(ws + 85516800);      // 64*704 bf16  = 88 KB

    k_segids <<<(N_SENT + 255) / 256, 256, 0, stream>>>(scope, seg);
    k_zero   <<<(NB * DH + 255) / 256, 256, 0, stream>>>(bagsum, NB * DH);
    k_prep_w <<<(DHP * DUP + 255) / 256, 256, 0, stream>>>(W1, W2, Wc, W1t, W2t, Wct);
    k_u2bf   <<<((long)N_SENT * DUP + 255) / 256, 256, 0, stream>>>(U, Ubf);
    k_fc1    <<<(N_SENT + 31) / 32, 256, 0, stream>>>(Ubf, W1t, b1, H);
    k_fc2_gate<<<(N_SENT + 63) / 64, 256, 0, stream>>>(H, W2t, b2, S, seg, bagsum);
    k_bagmean<<<(NB * DHP + 255) / 256, 256, 0, stream>>>(bagsum, scope, Bbf);
    k_cls    <<<(NB + 31) / 32, 256, 0, stream>>>(Bbf, Wct, bc, out);
}
